// AttentionBlock3D_80178449481825
// MI455X (gfx1250) — matrix operand order repaired and checked by reading, not compiled, not run on hardware
//
#include <hip/hip_runtime.h>
#include <hip/hip_bf16.h>

// ---------------- problem constants ----------------
#define CC    256
#define GRP   32
#define BB    2
#define DD    8
#define HH    48
#define WW    48
#define HWSZ  (HH * WW)          // 2304
#define DHW   (DD * HH * WW)     // 18432
#define NTOT  (BB * DHW)         // 36864
#define SCALE 0.17677669529663687f   // (C/8)^-0.5 = 1/sqrt(32)

typedef __bf16 bf16x16 __attribute__((ext_vector_type(16)));
typedef float  v8f     __attribute__((ext_vector_type(8)));
typedef unsigned int u32x4 __attribute__((ext_vector_type(4)));
typedef int          i32x8 __attribute__((ext_vector_type(8)));
typedef int          i32x4 __attribute__((ext_vector_type(4)));

struct Frag { union { bf16x16 v; unsigned int u[8]; }; };

__device__ __forceinline__ unsigned short f2bfu(float f) {
    unsigned int u = __float_as_uint(f);
    u += 0x7FFFu + ((u >> 16) & 1u);          // round-to-nearest-even
    return (unsigned short)(u >> 16);
}

// A-fragment (16x32 bf16) from row-major [m][k] bf16 matrix, ld in elements.
// Lane mapping: lanes 0-15 row m, K = 0..15; lanes 16-31 K = 16..31 (matches load_b).
__device__ __forceinline__ Frag load_a(const unsigned short* A, int ld, int m, int k0, int half) {
    Frag f;
#pragma unroll
    for (int j = 0; j < 8; ++j) {
        int kb = (half << 4) + (j << 1);
        f.u[j] = *reinterpret_cast<const unsigned int*>(A + (size_t)m * ld + k0 + kb);
    }
    return f;
}

// B-fragment (32x16 bf16) where B[k][n] = M[n][k] (per output column n, contiguous k).
// Lanes 0-15: K 0-15 (VGPR j -> K=2j,2j+1); lanes 16-31: K 16-31.
__device__ __forceinline__ Frag load_b(const unsigned short* M, int ld, int n, int k0, int half) {
    Frag f;
#pragma unroll
    for (int j = 0; j < 8; ++j) {
        int k = (half << 4) + (j << 1);
        f.u[j] = *reinterpret_cast<const unsigned int*>(M + (size_t)n * ld + k0 + k);
    }
    return f;
}

// LDS byte offset of a __shared__ object (generic LDS addresses carry the
// wave-relative offset in their low 32 bits; hardware adds LDS_BASE).
__device__ __forceinline__ unsigned lds_off(const void* p) {
    return (unsigned)(unsigned long long)p;
}

// ---- Tensor Data Mover: DMA a dense 2-D bf16 tile (tile0 x tile1, row stride
// stride0 elements) from global memory into LDS. D# built per ISA 8.3/8.4:
// group0 = {count=1 | lds_addr | global_addr | type=2}, group1 = dims/strides,
// groups 2/3 zero (2-D tensor; tile_dim2/3 = 0 -> unused). Tracked by TENSORcnt.
__device__ __forceinline__ void tdm_load_2d(unsigned lds_byte_off, const void* gptr,
                                            unsigned tile0, unsigned tile1,
                                            unsigned long long stride0) {
    unsigned long long ga = (unsigned long long)gptr;
    u32x4 g0;
    g0[0] = 1u;                                           // count=1, user mode
    g0[1] = lds_byte_off;                                 // lds_addr
    g0[2] = (unsigned)ga;                                 // global_addr[31:0]
    g0[3] = ((unsigned)(ga >> 32) & 0x01FFFFFFu) | (2u << 30);  // addr[56:32] | type=2
    i32x8 g1;
    g1[0] = 0x00010000;                                   // data_size=1 (2 bytes)
    g1[1] = (int)((tile0 & 0xFFFFu) << 16);               // tensor_dim0[15:0]
    g1[2] = (int)(((tile1 & 0xFFFFu) << 16) | (tile0 >> 16)); // dim0 hi | dim1 lo
    g1[3] = (int)((tile0 << 16) | (tile1 >> 16));         // dim1 hi | tile_dim0
    g1[4] = (int)(tile1 & 0xFFFFu);                       // tile_dim1 | tile_dim2=0
    g1[5] = (int)(stride0 & 0xFFFFFFFFull);               // dim0_stride[31:0]
    g1[6] = (int)((stride0 >> 32) & 0xFFFFull);           // dim0_stride[47:32] | dim1_stride=0
    g1[7] = 0;
    i32x4 gz = {0, 0, 0, 0};
    asm volatile("tensor_load_to_lds %0, %1, %2, %3"
                 :: "s"(g0), "s"(g1), "s"(gz), "s"(gz) : "memory");
}

// ---------------- 1: GroupNorm statistics ----------------
__global__ void gn_stats_kernel(const float* __restrict__ x, float* mu, float* rsig) {
    __shared__ float s1[256], s2[256];
    int bg = blockIdx.x;                       // b*32+g  -> (b*256+g*8)*DHW == bg*8*DHW
    const float* base = x + (size_t)bg * 8 * DHW;
    float a = 0.f, q = 0.f;
    for (int i = threadIdx.x; i < 8 * DHW; i += 256) {
        float v = base[i]; a += v; q += v * v;
    }
    s1[threadIdx.x] = a; s2[threadIdx.x] = q;
    __syncthreads();
    for (int st = 128; st > 0; st >>= 1) {
        if ((int)threadIdx.x < st) {
            s1[threadIdx.x] += s1[threadIdx.x + st];
            s2[threadIdx.x] += s2[threadIdx.x + st];
        }
        __syncthreads();
    }
    if (threadIdx.x == 0) {
        float inv = 1.0f / (8.0f * (float)DHW);
        float m = s1[0] * inv;
        float var = s2[0] * inv - m * m;
        mu[bg] = m;
        rsig[bg] = rsqrtf(var + 1e-5f);
    }
}

// ---------------- 2: weights f32 -> bf16 ----------------
__global__ void cvt_w_kernel(const float* __restrict__ qkvw, const float* __restrict__ projw,
                             unsigned short* qbf, unsigned short* pbf) {
    int i = blockIdx.x * blockDim.x + threadIdx.x;
    if (i < 768 * 256) qbf[i] = f2bfu(qkvw[i]);
    if (i < 256 * 256) pbf[i] = f2bfu(projw[i]);
}

// ---------------- 3: normalize + transpose to h[n][c] bf16 ----------------
__global__ void norm_tr_kernel(const float* __restrict__ x,
                               const float* __restrict__ nw, const float* __restrict__ nb,
                               const float* __restrict__ mu, const float* __restrict__ rsig,
                               unsigned short* __restrict__ ht) {
    __shared__ float tile[32][33];
    int b = blockIdx.z, cBase = blockIdx.y * 32, nBase = blockIdx.x * 32;
    int tx = threadIdx.x, ty = threadIdx.y;    // 32 x 8
#pragma unroll
    for (int i = 0; i < 4; ++i) {
        int c = cBase + ty + i * 8;
        float v = x[((size_t)b * CC + c) * DHW + nBase + tx];
        int g = b * GRP + (c >> 3);
        tile[ty + i * 8][tx] = (v - mu[g]) * rsig[g] * nw[c] + nb[c];
    }
    __syncthreads();
#pragma unroll
    for (int i = 0; i < 4; ++i) {
        int n = nBase + ty + i * 8;
        ht[((size_t)b * DHW + n) * CC + cBase + tx] = f2bfu(tile[tx][ty + i * 8]);
    }
}

// ---------------- 4: QKV GEMM (768x256) x h -> q[n][c]*scale, k[n][c], v[c][n] ----------------
__global__ void qkv_gemm_kernel(const unsigned short* __restrict__ wbf,
                                const unsigned short* __restrict__ ht,
                                const float* __restrict__ qkv_b,
                                unsigned short* __restrict__ qt,
                                unsigned short* __restrict__ kt,
                                unsigned short* __restrict__ vc) {
    int wave = blockIdx.x * 8 + (threadIdx.x >> 5);
    int lane = threadIdx.x & 31, half = lane >> 4, lq = lane & 15;
    int mt = wave % 48, nt = wave / 48;        // M tiles: 768/16, N tiles: 36864/16
    v8f acc = {};
#pragma unroll
    for (int ks = 0; ks < 8; ++ks) {
        Frag a = load_a(wbf, CC, mt * 16 + lq, ks * 32, half);
        Frag bfr = load_b(ht, CC, nt * 16 + lq, ks * 32, half);
        acc = __builtin_amdgcn_wmma_f32_16x16x32_bf16(false, a.v, false, bfr.v,
                                                      (short)0, acc, false, false);
    }
    int n_g = nt * 16 + lq;
    int b = n_g / DHW, sp = n_g % DHW;
    int o0 = mt * 16;
#pragma unroll
    for (int r = 0; r < 8; ++r) {
        int o = o0 + r + 8 * half;
        float d = acc[r] + qkv_b[o];
        if (o0 < 256)
            qt[(size_t)n_g * CC + o] = f2bfu(d * SCALE);
        else if (o0 < 512)
            kt[(size_t)n_g * CC + (o - 256)] = f2bfu(d);
        else
            vc[((size_t)b * CC + (o - 512)) * DHW + sp] = f2bfu(d);
    }
}

// ---------------- 5: fused attention per (b,d) slice ----------------
// Softmax is over the LAST axis only (48 keys of one H row) -> each 48-key block
// has a fully local softmax: no online rescaling needed. K/V blocks are staged
// in LDS once per workgroup via double-buffered TDM DMA (tensor_load_to_lds).
__global__ void __launch_bounds__(256, 1)
attn_kernel(const unsigned short* __restrict__ qt, const unsigned short* __restrict__ kt,
            const unsigned short* __restrict__ vc, unsigned short* __restrict__ ao) {
    int slice = blockIdx.y;                    // b*D+d
    int b = slice >> 3, d = slice & 7;
    int wv = threadIdx.x >> 5, lane = threadIdx.x & 31, half = lane >> 4, lq = lane & 15;
    int q0 = blockIdx.x * 128 + wv * 16;       // wave's 16 queries
    int sliceN = b * DHW + d * HWSZ;           // base n-index of this depth slice

    __shared__ unsigned short kbuf[2][48 * 256];   // K block: 48 keys x 256 c   (2x24KB)
    __shared__ unsigned short vbuf[2][256 * 48];   // V block: 256 c x 48 keys   (2x24KB)
    __shared__ unsigned short plds[8][16 * 64];    // per-wave P tile (cols 48..63 zero)

    // q fragments (A-layout), loop invariant over H blocks
    Frag qf[8];
#pragma unroll
    for (int ks = 0; ks < 8; ++ks)
        qf[ks] = load_a(qt, CC, sliceN + q0 + lq, ks * 32, half);

    v8f o_acc[16];
#pragma unroll
    for (int ct = 0; ct < 16; ++ct) o_acc[ct] = (v8f){};

    const unsigned short* kslice = kt + (size_t)sliceN * CC;                  // [key][c]
    const unsigned short* vslice = vc + (size_t)b * CC * DHW + d * HWSZ;      // [c][key], ld=DHW

    // prime the pipeline: DMA block 0 (wave0 -> K, wave1 -> V)
    if (wv == 0) tdm_load_2d(lds_off(kbuf[0]), kslice, 256, 48, 256);
    if (wv == 1) tdm_load_2d(lds_off(vbuf[0]), vslice, 48, 256, DHW);

    for (int hb = 0; hb < 48; ++hb) {
        int cur = hb & 1;
        if (wv < 2) __builtin_amdgcn_s_wait_tensorcnt(0);   // this block's DMA done
        __syncthreads();
        if (hb + 1 < 48) {                                  // overlap: DMA next block
            int nxt = cur ^ 1;
            if (wv == 0) tdm_load_2d(lds_off(kbuf[nxt]),
                                     kslice + (size_t)(hb + 1) * 48 * CC, 256, 48, 256);
            if (wv == 1) tdm_load_2d(lds_off(vbuf[nxt]),
                                     vslice + (hb + 1) * 48, 48, 256, DHW);
        }

        // ---- S = q . k^T over this H block (16 queries x 48 keys), K from LDS ----
        v8f s[3];
#pragma unroll
        for (int nt = 0; nt < 3; ++nt) {
            s[nt] = (v8f){};
            int keyl = nt * 16 + lq;
#pragma unroll
            for (int ks = 0; ks < 8; ++ks) {
                Frag kf = load_b(kbuf[cur], CC, keyl, ks * 32, half);
                s[nt] = __builtin_amdgcn_wmma_f32_16x16x32_bf16(false, qf[ks].v, false, kf.v,
                                                                (short)0, s[nt], false, false);
            }
        }

        // ---- per-row softmax over the 48 keys (row = VGPR r, cols across 16 lanes x 3 tiles)
#pragma unroll
        for (int r = 0; r < 8; ++r) {
            float mx = fmaxf(fmaxf(s[0][r], s[1][r]), s[2][r]);
#pragma unroll
            for (int m = 1; m < 16; m <<= 1) mx = fmaxf(mx, __shfl_xor(mx, m, 32));
            float e0 = __expf(s[0][r] - mx);
            float e1 = __expf(s[1][r] - mx);
            float e2 = __expf(s[2][r] - mx);
            float sum = e0 + e1 + e2;
#pragma unroll
            for (int m = 1; m < 16; m <<= 1) sum += __shfl_xor(sum, m, 32);
            float inv = 1.0f / sum;
            s[0][r] = e0 * inv; s[1][r] = e1 * inv; s[2][r] = e2 * inv;
        }

        // ---- re-layout P: C-layout regs -> LDS -> A-layout fragments (K padded 48->64)
#pragma unroll
        for (int nt = 0; nt < 3; ++nt)
#pragma unroll
            for (int r = 0; r < 8; ++r)
                plds[wv][(r + 8 * half) * 64 + nt * 16 + lq] = f2bfu(s[nt][r]);
#pragma unroll
        for (int r = 0; r < 8; ++r)
            plds[wv][(r + 8 * half) * 64 + 48 + lq] = 0;    // zero pad cols 48..63

        Frag pf0, pf1;
#pragma unroll
        for (int j = 0; j < 8; ++j) {
            int kb = (half << 4) + (j << 1);
            pf0.u[j] = *reinterpret_cast<const unsigned int*>(&plds[wv][lq * 64 + kb]);
            pf1.u[j] = *reinterpret_cast<const unsigned int*>(&plds[wv][lq * 64 + 32 + kb]);
        }

        // ---- O += P @ V_block (K = 64 with zero pad; V block from LDS [c][48]) ----
#pragma unroll
        for (int ct = 0; ct < 16; ++ct) {
            const unsigned short* vrow = &vbuf[cur][(ct * 16 + lq) * 48];
            Frag vf;
#pragma unroll
            for (int j = 0; j < 8; ++j) {
                int k = (half << 4) + (j << 1);
                vf.u[j] = *reinterpret_cast<const unsigned int*>(vrow + k);
            }
            o_acc[ct] = __builtin_amdgcn_wmma_f32_16x16x32_bf16(false, pf0.v, false, vf.v,
                                                                (short)0, o_acc[ct], false, false);
#pragma unroll
            for (int j = 0; j < 8; ++j) {
                int k = 32 + (half << 4) + (j << 1);
                vf.u[j] = (k < 48) ? *reinterpret_cast<const unsigned int*>(vrow + k) : 0u;
            }
            o_acc[ct] = __builtin_amdgcn_wmma_f32_16x16x32_bf16(false, pf1.v, false, vf.v,
                                                                (short)0, o_acc[ct], false, false);
        }
        __syncthreads();   // all consumers done before this buffer is re-filled
    }

    // store attention output as ao[n][c] bf16 (proj GEMM wants per-n contiguous c)
#pragma unroll
    for (int ct = 0; ct < 16; ++ct)
#pragma unroll
        for (int r = 0; r < 8; ++r) {
            int qrow = q0 + r + 8 * half;
            ao[(size_t)(sliceN + qrow) * CC + ct * 16 + lq] = f2bfu(o_acc[ct][r]);
        }
}

// ---------------- 6: proj GEMM + bias + residual ----------------
__global__ void proj_gemm_kernel(const unsigned short* __restrict__ wbf,
                                 const unsigned short* __restrict__ ao,
                                 const float* __restrict__ proj_b,
                                 const float* __restrict__ x,
                                 float* __restrict__ out) {
    int wave = blockIdx.x * 8 + (threadIdx.x >> 5);
    int lane = threadIdx.x & 31, half = lane >> 4, lq = lane & 15;
    int mt = wave % 16, nt = wave / 16;        // M: 256/16, N: 36864/16
    v8f acc = {};
#pragma unroll
    for (int ks = 0; ks < 8; ++ks) {
        Frag a = load_a(wbf, CC, mt * 16 + lq, ks * 32, half);
        Frag bfr = load_b(ao, CC, nt * 16 + lq, ks * 32, half);
        acc = __builtin_amdgcn_wmma_f32_16x16x32_bf16(false, a.v, false, bfr.v,
                                                      (short)0, acc, false, false);
    }
    int n_g = nt * 16 + lq;
    int b = n_g / DHW, sp = n_g % DHW;
#pragma unroll
    for (int r = 0; r < 8; ++r) {
        int o = mt * 16 + r + 8 * half;
        size_t idx = ((size_t)b * CC + o) * DHW + sp;
        out[idx] = x[idx] + proj_b[o] + acc[r];
    }
}

// ---------------- launcher ----------------
extern "C" void kernel_launch(void* const* d_in, const int* in_sizes, int n_in,
                              void* d_out, int out_size, void* d_ws, size_t ws_size,
                              hipStream_t stream) {
    const float* x      = (const float*)d_in[0];
    const float* norm_w = (const float*)d_in[1];
    const float* norm_b = (const float*)d_in[2];
    const float* qkv_w  = (const float*)d_in[3];
    const float* qkv_b  = (const float*)d_in[4];
    const float* proj_w = (const float*)d_in[5];
    const float* proj_b = (const float*)d_in[6];
    float* out = (float*)d_out;

    char* ws = (char*)d_ws;
    float* mu   = (float*)(ws);
    float* rsig = (float*)(ws + 256);
    unsigned short* qkvw_bf = (unsigned short*)(ws + 512);
    unsigned short* projw_bf = (unsigned short*)(ws + 512 + 768 * 256 * 2);
    size_t off = 512 + 768 * 256 * 2 + 256 * 256 * 2;     // 524800, 256B aligned
    const size_t act = (size_t)NTOT * CC * 2;             // 18.9 MB each
    unsigned short* ht = (unsigned short*)(ws + off); off += act;
    unsigned short* qt = (unsigned short*)(ws + off); off += act;
    unsigned short* kt = (unsigned short*)(ws + off); off += act;
    unsigned short* vc = (unsigned short*)(ws + off); off += act;
    unsigned short* ao = (unsigned short*)(ws + off); off += act;   // ~94.9 MB total

    gn_stats_kernel<<<BB * GRP, 256, 0, stream>>>(x, mu, rsig);
    cvt_w_kernel<<<(768 * 256) / 256, 256, 0, stream>>>(qkv_w, proj_w, qkvw_bf, projw_bf);
    norm_tr_kernel<<<dim3(DHW / 32, CC / 32, BB), dim3(32, 8), 0, stream>>>(
        x, norm_w, norm_b, mu, rsig, ht);
    qkv_gemm_kernel<<<(48 * (NTOT / 16)) / 8, 256, 0, stream>>>(
        qkvw_bf, ht, qkv_b, qt, kt, vc);
    attn_kernel<<<dim3(HWSZ / 128, BB * DD), 256, 0, stream>>>(qt, kt, vc, ao);
    proj_gemm_kernel<<<(16 * (NTOT / 16)) / 8, 256, 0, stream>>>(
        projw_bf, ao, proj_b, x, out);
}
